// GridNeRFRenderer_55465207661096
// MI455X (gfx1250) — matrix operand order repaired and checked by reading, not compiled, not run on hardware
//
#include <hip/hip_runtime.h>
#include <math.h>

// ---------------------------------------------------------------------------
// NeRF hierarchical sampling + volume rendering for MI455X (gfx1250, wave32).
// Block = 64 threads (2 waves). Each wave owns 16 rays.
//  Phase 0: async global->LDS staging of coarse_t (CDNA5 async-tensor path)
//  Phase 1: CDF via WMMA (cumsum == GEMM with constant triangular-ones B)
//  Phase 2: inverse-CDF sampling (binary search on LDS cdf, fast rcp)
//  Phase 3: bitonic sort of 128 fine t's per ray (LDS)
//  Phase 4: merge-by-ranks with sorted coarse t's -> 192 sorted samples
//  Phase 5: volume rendering split-scan (2 lanes per ray + shfl_xor stitch)
// ---------------------------------------------------------------------------

typedef __attribute__((ext_vector_type(16))) _Float16 v16h;
typedef __attribute__((ext_vector_type(8)))  float    v8f;
typedef __attribute__((ext_vector_type(4)))  int      v4i;

#define NC 64
#define NF 128
#define NS 192
#define RAYS_PER_WAVE  16
#define RAYS_PER_BLOCK 32
#define THREADS        64

#define AS1 __attribute__((address_space(1)))
#define AS3 __attribute__((address_space(3)))

#if defined(__gfx1250__) && __has_builtin(__builtin_amdgcn_global_load_async_to_lds_b128)
#define HAVE_ASYNC_LDS 1
#endif

static __device__ __forceinline__ void wait_async_zero() {
#if defined(HAVE_ASYNC_LDS)
#if __has_builtin(__builtin_amdgcn_s_wait_asynccnt)
  __builtin_amdgcn_s_wait_asynccnt(0);
#else
  asm volatile("s_wait_asynccnt 0" ::: "memory");
#endif
#endif
}

// fast reciprocal: v_rcp_f32 (1 ulp) -- both uses are well-conditioned
static __device__ __forceinline__ float fast_rcp(float x) {
#if __has_builtin(__builtin_amdgcn_rcpf)
  return __builtin_amdgcn_rcpf(x);
#else
  return 1.0f / x;
#endif
}

__global__ __launch_bounds__(THREADS)
void nerf_render_kernel(const float* __restrict__ dirs,
                        const float* __restrict__ cw,     // coarse_weights [N,64]
                        const float* __restrict__ ct,     // coarse_t_vals  [N,64]
                        const float* __restrict__ uu,     // u              [N,128]
                        const float* __restrict__ colors, // [N,192,3]
                        const float* __restrict__ dens,   // [N,192]
                        const float* __restrict__ bg,     // [3]
                        float* __restrict__ out)          // [N,5]
{
  __shared__ float sCoarse[RAYS_PER_BLOCK][NC];
  __shared__ float sCdf   [RAYS_PER_BLOCK][NC + 2];  // indices 0..64 used
  __shared__ float sFine  [RAYS_PER_BLOCK][NF];
  __shared__ float sMerged[RAYS_PER_BLOCK][NS];
  __shared__ float sInv   [RAYS_PER_BLOCK];

  const int wl   = threadIdx.x & 31;   // lane in wave32
  const int wave = threadIdx.x >> 5;   // 0 or 1
  const int rwb  = wave * RAYS_PER_WAVE;                 // wave's first row in block
  const int rb   = blockIdx.x * RAYS_PER_BLOCK + rwb;    // wave's first global ray

  // ---- Phase 0: stage this wave's 16 contiguous rows of coarse_t (4KB) into LDS
  {
    const char* gsrc = (const char*)(ct + (size_t)rb * NC);
    char*       ldst = (char*)&sCoarse[rwb][0];
#if defined(HAVE_ASYNC_LDS)
#pragma unroll
    for (int it = 0; it < 8; ++it) {
      int off = (wl + 32 * it) * 16;
      __builtin_amdgcn_global_load_async_to_lds_b128(
          (AS1 v4i*)(size_t)(gsrc + off),
          (AS3 v4i*)(unsigned)(size_t)(ldst + off), 0, 0);
    }
#else
#pragma unroll
    for (int it = 0; it < 8; ++it) {
      int off = wl + 32 * it;
      ((float4*)ldst)[off] = ((const float4*)gsrc)[off];
    }
#endif
  }

  // ---- Phase 1: raw inclusive cumsum of (w + 1e-5) via WMMA: D = A(16x64) x L^T
  // Output tile nt covers cdf columns 16*nt..16*nt+15; K split into two 32-chunks.
  {
    const int m  = wl & 15;
    const int hl = wl >> 4;
    const float* Wrow = cw + (size_t)(rb + m) * NC;
    v16h a0, a1;
#pragma unroll
    for (int v = 0; v < 8; ++v) {
      int kb = ((v < 4) ? (2 * v) : (16 + 2 * (v - 4))) + 8 * hl; // documented A16x32 layout
      a0[2 * v]     = (_Float16)(Wrow[kb]      + 1e-5f);
      a0[2 * v + 1] = (_Float16)(Wrow[kb + 1]  + 1e-5f);
      a1[2 * v]     = (_Float16)(Wrow[kb + 32] + 1e-5f);
      a1[2 * v + 1] = (_Float16)(Wrow[kb + 33] + 1e-5f);
    }
    // B(32x16): N column = lane&15, element e = K row in table order. Triangular / ones.
    v16h bLo, bHi, bOne;
#pragma unroll
    for (int e = 0; e < 16; ++e) {
      int kk = e + ((e < 8) ? 0 : 8) + 8 * hl;
      bLo[e]  = (_Float16)((kk <= (wl & 15))      ? 1.0f : 0.0f);  // K <= N
      bHi[e]  = (_Float16)((kk <= (wl & 15) + 16) ? 1.0f : 0.0f);  // K <= N+16
      bOne[e] = (_Float16)1.0f;
    }
    v8f acc0 = {}, acc1 = {}, acc2 = {}, acc3 = {};
    acc0 = __builtin_amdgcn_wmma_f32_16x16x32_f16(false, a0, false, bLo,  (short)0, acc0, false, false);
    acc1 = __builtin_amdgcn_wmma_f32_16x16x32_f16(false, a0, false, bHi,  (short)0, acc1, false, false);
    acc2 = __builtin_amdgcn_wmma_f32_16x16x32_f16(false, a0, false, bOne, (short)0, acc2, false, false);
    acc2 = __builtin_amdgcn_wmma_f32_16x16x32_f16(false, a1, false, bLo,  (short)0, acc2, false, false);
    acc3 = __builtin_amdgcn_wmma_f32_16x16x32_f16(false, a0, false, bOne, (short)0, acc3, false, false);
    acc3 = __builtin_amdgcn_wmma_f32_16x16x32_f16(false, a1, false, bHi,  (short)0, acc3, false, false);

    const int n = wl & 15;
#pragma unroll
    for (int r = 0; r < 8; ++r) {   // D layout: VGPR r -> M = r + 8*hl, N = lane&15
      int mm = rwb + r + 8 * hl;
      sCdf[mm][ 0 + n + 1] = acc0[r];
      sCdf[mm][16 + n + 1] = acc1[r];
      sCdf[mm][32 + n + 1] = acc2[r];
      sCdf[mm][48 + n + 1] = acc3[r];
    }
  }
  __syncthreads();

  // normalize: cdf[0]=0, cdf[j]=C[j-1]/C[63]  (total >= 64e-5, rcp is safe)
  if (wl < RAYS_PER_WAVE) sInv[rwb + wl] = fast_rcp(sCdf[rwb + wl][NC]);
  __syncthreads();
  for (int q = wl; q < RAYS_PER_WAVE * (NC + 1); q += 32) {
    int m = rwb + q / (NC + 1);
    int j = q % (NC + 1);
    sCdf[m][j] = (j == 0) ? 0.0f : sCdf[m][j] * sInv[m];
  }
  wait_async_zero();   // coarse_t needed from here on
  __syncthreads();

  // ---- Phase 2: inverse-CDF sampling (searchsorted 'right' + lerp)
  for (int q = wl; q < RAYS_PER_WAVE * NF; q += 32) {
    int m = rwb + (q >> 7);
    int f = q & (NF - 1);
    float u = uu[(size_t)(blockIdx.x * RAYS_PER_BLOCK + m) * NF + f];
    int lo = 0, hi = NC + 1;
    while (lo < hi) { int mid = (lo + hi) >> 1; if (sCdf[m][mid] <= u) lo = mid + 1; else hi = mid; }
    int below = lo - 1; below = below < 0 ? 0 : (below > NC - 1 ? NC - 1 : below);
    int above = lo;     above = above < 0 ? 0 : (above > NC - 1 ? NC - 1 : above);
    float cb = sCdf[m][below], ca = sCdf[m][above];
    float denom = ca - cb; if (denom < 1e-5f) denom = 1.0f;
    float t  = (u - cb) * fast_rcp(denom);           // denom >= 1e-5 here
    float tb = sCoarse[m][below], ta = sCoarse[m][above];
    sFine[m][f] = tb + t * (ta - tb);
  }
  __syncthreads();

  // ---- Phase 3: bitonic sort of 128 fine t's per ray (16 rays in lockstep)
  for (int k = 2; k <= NF; k <<= 1) {
    for (int j = k >> 1; j > 0; j >>= 1) {
      for (int it = 0; it < (RAYS_PER_WAVE * (NF / 2)) / 32; ++it) {
        int p  = wl + 32 * it;
        int m  = rwb + (p >> 6);
        int q  = p & 63;
        int i  = ((q & ~(j - 1)) << 1) | (q & (j - 1));
        int ix = i | j;
        bool asc = ((i & k) == 0);
        float a = sFine[m][i], b = sFine[m][ix];
        float lo2 = fminf(a, b), hi2 = fmaxf(a, b);
        sFine[m][i]  = asc ? lo2 : hi2;
        sFine[m][ix] = asc ? hi2 : lo2;
      }
      __syncthreads();
    }
  }

  // ---- Phase 4: merge coarse(64, sorted) + fine(128, sorted) by ranks
  for (int q = wl; q < RAYS_PER_WAVE * NS; q += 32) {
    int m, pos; float val;
    if (q < RAYS_PER_WAVE * NC) {           // coarse element
      m = rwb + (q >> 6);
      int i = q & (NC - 1);
      val = sCoarse[m][i];
      int lo = 0, hi = NF;                  // count fine < val
      while (lo < hi) { int mid = (lo + hi) >> 1; if (sFine[m][mid] < val) lo = mid + 1; else hi = mid; }
      pos = i + lo;
    } else {                                 // fine element
      int qq = q - RAYS_PER_WAVE * NC;
      m = rwb + (qq >> 7);
      int j = qq & (NF - 1);
      val = sFine[m][j];
      int lo = 0, hi = NC;                  // count coarse <= val
      while (lo < hi) { int mid = (lo + hi) >> 1; if (sCoarse[m][mid] <= val) lo = mid + 1; else hi = mid; }
      pos = j + lo;
    }
    sMerged[m][pos] = val;
  }
  __syncthreads();

  // ---- Phase 5: volume rendering. 2 lanes per ray; two-pass split scan.
  {
    const int m    = wl & 15;
    const int p    = wl >> 4;              // half: 0 -> s[0,96), 1 -> s[96,192)
    const int mrow = rwb + m;
    const int ray  = blockIdx.x * RAYS_PER_BLOCK + mrow;
    float dx = dirs[3 * ray], dy = dirs[3 * ray + 1], dz = dirs[3 * ray + 2];
    float nrm = sqrtf(dx * dx + dy * dy + dz * dz);
    const float* drow = dens   + (size_t)ray * NS;
    const float* crow = colors + (size_t)ray * NS * 3;

    // pass 1: alphas (cached into dead sFine/sCoarse regions) + local product
    float P = 1.0f;
    const int s0 = p * (NS / 2);
    for (int qs = 0; qs < NS / 2; ++qs) {
      int s = s0 + qs;
      float t0v  = sMerged[mrow][s];
      float dist = (s == NS - 1) ? 1e10f : (sMerged[mrow][s + 1] - t0v);
      dist *= nrm;
      float aL = 1.0f - __expf(-drow[s] * dist);
      if (s < NF) sFine[mrow][s] = aL; else sCoarse[mrow][s - NF] = aL;
      P *= (1.0f - aL + 1e-10f);
    }
    float Pother = __shfl_xor(P, 16, 32);
    float T = (p == 0) ? 1.0f : Pother;

    // pass 2: weighted accumulation with running transmittance
    float sw = 0.f, swt = 0.f, c0 = 0.f, c1 = 0.f, c2 = 0.f;
    for (int qs = 0; qs < NS / 2; ++qs) {
      int s = s0 + qs;
      float aL = (s < NF) ? sFine[mrow][s] : sCoarse[mrow][s - NF];
      float w  = aL * T;
      T *= (1.0f - aL + 1e-10f);
      float tv = sMerged[mrow][s];
      sw  += w;
      swt += w * tv;
      c0  += w * crow[3 * s + 0];
      c1  += w * crow[3 * s + 1];
      c2  += w * crow[3 * s + 2];
    }
    sw  += __shfl_xor(sw, 16, 32);
    swt += __shfl_xor(swt, 16, 32);
    c0  += __shfl_xor(c0, 16, 32);
    c1  += __shfl_xor(c1, 16, 32);
    c2  += __shfl_xor(c2, 16, 32);
    if (p == 0) {
      float acc = sw;
      float* o = out + (size_t)ray * 5;
      o[0] = c0 + (1.0f - acc) * bg[0];
      o[1] = c1 + (1.0f - acc) * bg[1];
      o[2] = c2 + (1.0f - acc) * bg[2];
      o[3] = swt;
      o[4] = acc;
    }
  }
}

extern "C" void kernel_launch(void* const* d_in, const int* in_sizes, int n_in,
                              void* d_out, int out_size, void* d_ws, size_t ws_size,
                              hipStream_t stream) {
  (void)n_in; (void)out_size; (void)d_ws; (void)ws_size;
  // input order: ray_origins, ray_directions, coarse_weights, coarse_t_vals,
  //              u, colors, densities, background_color
  const float* dirs   = (const float*)d_in[1];
  const float* cw     = (const float*)d_in[2];
  const float* ct     = (const float*)d_in[3];
  const float* uu     = (const float*)d_in[4];
  const float* colors = (const float*)d_in[5];
  const float* dens   = (const float*)d_in[6];
  const float* bg     = (const float*)d_in[7];
  float* out = (float*)d_out;

  int nRays = in_sizes[2] / NC;            // 65536
  int grid  = nRays / RAYS_PER_BLOCK;      // 2048
  hipLaunchKernelGGL(nerf_render_kernel, dim3(grid), dim3(THREADS), 0, stream,
                     dirs, cw, ct, uu, colors, dens, bg, out);
}
